// LlamaAttention_29180007809075
// MI455X (gfx1250) — matrix operand order repaired and checked by reading, not compiled, not run on hardware
//
#include <hip/hip_runtime.h>
#include <hip/hip_bf16.h>
#include <stdint.h>

#define BB   2
#define SS   2048
#define HIDD 2048
#define NH   16
#define NKH  4
#define HD   128
#define NREP 4
#define BSROWS (BB * SS)   // 4096

static const float SCALE_F = 0.08838834764831845f; // 1/sqrt(128)

typedef __attribute__((ext_vector_type(16))) __bf16 bf16x16;
typedef __attribute__((ext_vector_type(8)))  __bf16 bf16x8;
typedef __attribute__((ext_vector_type(8)))  float  f32x8;

__device__ __forceinline__ __bf16 f2bf(float f) {
  unsigned u = __builtin_bit_cast(unsigned, f);
  unsigned r = (u + 0x7fffu + ((u >> 16) & 1u)) >> 16; // RNE
  return __builtin_bit_cast(__bf16, (unsigned short)r);
}

// Build a 16-element bf16 A/B fragment from two 16-byte chunks.
__device__ __forceinline__ bf16x16 mk_frag(const __bf16* lo, const __bf16* hi) {
  bf16x8 a = *(const bf16x8*)lo;
  bf16x8 b = *(const bf16x8*)hi;
  return __builtin_shufflevector(a, b, 0,1,2,3,4,5,6,7,8,9,10,11,12,13,14,15);
}

// CDNA5 async copy: 16 bytes per lane, global -> LDS, tracked by ASYNCcnt.
// GVS mode: lds_addr VGPR (u32, wave-relative LDS byte address),
//           off VGPR (u32 byte offset), base SGPR pair.
__device__ __forceinline__ void async_copy16(unsigned lds_addr, const void* base, unsigned off) {
  asm volatile("global_load_async_to_lds_b128 %0, %1, %2"
               :: "v"(lds_addr), "v"(off), "s"(base)
               : "memory");
}
__device__ __forceinline__ void wait_async_le4() {
  asm volatile("s_wait_asynccnt 0x4" ::: "memory");
}
__device__ __forceinline__ void wait_async_0() {
  asm volatile("s_wait_asynccnt 0x0" ::: "memory");
}

// ---------------- elementwise conversion kernels ----------------

__global__ void cvt_bf16_kernel(const float* __restrict__ in, __bf16* __restrict__ out, int n) {
  int i = blockIdx.x * blockDim.x + threadIdx.x;
  if (i < n) out[i] = f2bf(in[i]);
}

// in: [K][N] f32 row-major  ->  out: [N][K] bf16 row-major (transposed)
__global__ void cvt_bf16_t_kernel(const float* __restrict__ in, __bf16* __restrict__ out,
                                  int K, int N) {
  int i = blockIdx.x * blockDim.x + threadIdx.x;
  if (i >= K * N) return;
  int k = i / N, n = i - k * N;
  out[(size_t)n * K + k] = f2bf(in[i]);
}

// ---------------- bf16 GEMM: C[M,N] = A[M,K] @ Bt[N,K]^T ----------------
// Block: 256 threads = 8 waves; tile 128(M) x 128(N); K-step 32.
// Double-buffered LDS staged with global_load_async_to_lds_b128 (ASYNCcnt).
// Each wave: 32(M) x 64(N) = 2x4 WMMA accumulators -> 8 WMMAs per K-step.

__global__ __launch_bounds__(256) void gemm_bf16_kernel(
    const __bf16* __restrict__ A, const __bf16* __restrict__ Bt,
    float* __restrict__ C, int M, int N, int K)
{
  __shared__ __align__(16) __bf16 As[2][128 * 32];   // m-major [128][32]
  __shared__ __align__(16) __bf16 Bs[2][128 * 32];   // n-major [128][32]
  const int tid  = threadIdx.x;
  const int lane = tid & 31;
  const int wave = tid >> 5;
  const int wm = wave >> 1;   // 0..3 : 32-row slab
  const int wn = wave & 1;    // 0..1 : 64-col slab
  const int m0 = blockIdx.y * 128;
  const int n0 = blockIdx.x * 128;
  const int lm = lane & 15;
  const int lk = lane >> 4;

  // Staging decomposition: 512 16B chunks per matrix per stage, 256 threads
  // -> each thread moves rows r0 and r0+64, one 16B (8-elem) chunk each.
  const int r0 = tid >> 2;
  const int c0 = (tid & 3) * 8;

  const unsigned aLds[2] = { (unsigned)(size_t)&As[0][0], (unsigned)(size_t)&As[1][0] };
  const unsigned bLds[2] = { (unsigned)(size_t)&Bs[0][0], (unsigned)(size_t)&Bs[1][0] };
  const unsigned ldsOff0 = (unsigned)((r0 * 32 + c0) * 2);
  const unsigned ldsOff1 = (unsigned)(((r0 + 64) * 32 + c0) * 2);

  f32x8 acc[2][4] = {};
  const int nk = K >> 5;

  auto stage = [&](int k0, int buf) {
    unsigned ga0 = (unsigned)(((size_t)(m0 + r0)      * K + k0 + c0) * 2);
    unsigned ga1 = (unsigned)(((size_t)(m0 + r0 + 64) * K + k0 + c0) * 2);
    async_copy16(aLds[buf] + ldsOff0, A, ga0);
    async_copy16(aLds[buf] + ldsOff1, A, ga1);
    unsigned gb0 = (unsigned)(((size_t)(n0 + r0)      * K + k0 + c0) * 2);
    unsigned gb1 = (unsigned)(((size_t)(n0 + r0 + 64) * K + k0 + c0) * 2);
    async_copy16(bLds[buf] + ldsOff0, Bt, gb0);
    async_copy16(bLds[buf] + ldsOff1, Bt, gb1);
  };

  stage(0, 0);

  for (int ks = 0; ks < nk; ++ks) {
    const int buf = ks & 1;
    if (ks + 1 < nk) {
      stage((ks + 1) * 32, buf ^ 1); // prefetch next slab into other buffer
      wait_async_le4();              // oldest 4 (this slab) have landed
    } else {
      wait_async_0();
    }
    __syncthreads();

    bf16x16 afr[2], bfr[4];
#pragma unroll
    for (int mt = 0; mt < 2; ++mt) {
      const __bf16* p = &As[buf][(wm * 32 + mt * 16 + lm) * 32 + lk * 8];
      afr[mt] = mk_frag(p, p + 16);   // A 16x32: K halves {lk*8..+8} and {+16..}
    }
#pragma unroll
    for (int nt = 0; nt < 4; ++nt) {
      const __bf16* p = &Bs[buf][(wn * 64 + nt * 16 + lm) * 32 + lk * 8];
      bfr[nt] = mk_frag(p, p + 16);   // B 32x16: K halves {lk*8..+8} and {+16..}
    }
#pragma unroll
    for (int mt = 0; mt < 2; ++mt)
#pragma unroll
      for (int nt = 0; nt < 4; ++nt)
        acc[mt][nt] = __builtin_amdgcn_wmma_f32_16x16x32_bf16(
            false, afr[mt], false, bfr[nt], (short)0, acc[mt][nt], false, false);
    __syncthreads();
  }

  const int mrowoff = lk * 8;
#pragma unroll
  for (int mt = 0; mt < 2; ++mt)
#pragma unroll
    for (int nt = 0; nt < 4; ++nt) {
      int n = n0 + wn * 64 + nt * 16 + lm;
#pragma unroll
      for (int r = 0; r < 8; ++r) {
        int m = m0 + wm * 32 + mt * 16 + mrowoff + r;
        C[(size_t)m * N + n] = acc[mt][nt][r];
      }
    }
}

// ---------------- RoPE: f32 [B,S,heads*D] -> bf16 [B,heads,S,D] ----------------

__global__ void rope_kernel(const float* __restrict__ in, const float* __restrict__ cosb,
                            const float* __restrict__ sinb, __bf16* __restrict__ out,
                            int heads) {
  size_t idx = (size_t)blockIdx.x * blockDim.x + threadIdx.x;
  size_t total = (size_t)BB * SS * heads * HD;
  if (idx >= total) return;
  int d = (int)(idx & (HD - 1));
  int h = (int)((idx >> 7) % heads);
  int s = (int)((idx / ((size_t)HD * heads)) % SS);
  int b = (int)(idx / ((size_t)HD * heads * SS));
  size_t rowbase = ((size_t)(b * SS + s)) * heads * HD + (size_t)h * HD;
  float v = in[rowbase + d];
  float other = (d < HD / 2) ? -in[rowbase + d + HD / 2] : in[rowbase + d - HD / 2];
  size_t cs = ((size_t)(b * SS + s)) * HD + d;
  float r = v * cosb[cs] + other * sinb[cs];
  out[(((size_t)(b * heads + h)) * SS + s) * HD + d] = f2bf(r);
}

// ---------------- V reformat: f32 [B,S,KH*D] -> bf16 [B,KH,D,S] (d-major) ----------------

__global__ void v_transpose_kernel(const float* __restrict__ in, __bf16* __restrict__ out) {
  size_t idx = (size_t)blockIdx.x * blockDim.x + threadIdx.x;
  size_t total = (size_t)BB * SS * NKH * HD;
  if (idx >= total) return;
  int d  = (int)(idx & (HD - 1));
  int kh = (int)((idx >> 7) & (NKH - 1));
  int s  = (int)((idx / ((size_t)HD * NKH)) % SS);
  int b  = (int)(idx / ((size_t)HD * NKH * SS));
  out[(((size_t)(b * NKH + kh)) * HD + d) * SS + s] = f2bf(in[idx]);
}

// ---------------- causal GQA flash attention ----------------
// Grid: (S/128, NH, B). 256 threads = 8 waves; each wave owns a 16-query tile.
// Q: bf16 [B,NH,S,D]; K: bf16 [B,NKH,S,D]; Vt: bf16 [B,NKH,D,S]; O: bf16 [B*S, NH*D].

__global__ __launch_bounds__(256) void flash_attn_kernel(
    const __bf16* __restrict__ Q, const __bf16* __restrict__ Kc,
    const __bf16* __restrict__ Vt, __bf16* __restrict__ O)
{
  __shared__ __align__(16) __bf16 plds[8][16 * 32]; // per-wave P staging
  const int lane = threadIdx.x & 31;
  const int wave = threadIdx.x >> 5;
  const int h  = blockIdx.y;
  const int b  = blockIdx.z;
  const int q0 = blockIdx.x * 128 + wave * 16;
  const int kh = h / NREP;
  const int lm = lane & 15;
  const int lk = lane >> 4;
  const int mrowoff = lk * 8;

  const __bf16* Qb = Q  + ((size_t)(b * NH  + h )) * SS * HD;
  const __bf16* Kb = Kc + ((size_t)(b * NKH + kh)) * SS * HD;
  const __bf16* Vb = Vt + ((size_t)(b * NKH + kh)) * HD * SS;

  // Preload Q A-fragments for all 4 K-steps over D=128.
  bf16x16 qfr[4];
#pragma unroll
  for (int ks = 0; ks < 4; ++ks) {
    const __bf16* p = Qb + (size_t)(q0 + lm) * HD + ks * 32 + lk * 8;
    qfr[ks] = mk_frag(p, p + 16);
  }

  f32x8 oacc[8] = {};
  float mrun[8], lrun[8];
#pragma unroll
  for (int r = 0; r < 8; ++r) { mrun[r] = -3.0e38f; lrun[r] = 0.0f; }

  __bf16* pl = plds[wave];

  for (int k0 = 0; k0 < q0 + 16; k0 += 32) {
    // S = Q @ K^T for 32 keys: two 16-key n-tiles, 4 K-steps each.
    f32x8 s0 = {}, s1 = {};
#pragma unroll
    for (int ks = 0; ks < 4; ++ks) {
      const __bf16* p0 = Kb + (size_t)(k0 + lm) * HD + ks * 32 + lk * 8;
      bf16x16 b0 = mk_frag(p0, p0 + 16);
      s0 = __builtin_amdgcn_wmma_f32_16x16x32_bf16(false, qfr[ks], false, b0, (short)0, s0, false, false);
      const __bf16* p1 = Kb + (size_t)(k0 + 16 + lm) * HD + ks * 32 + lk * 8;
      bf16x16 b1 = mk_frag(p1, p1 + 16);
      s1 = __builtin_amdgcn_wmma_f32_16x16x32_bf16(false, qfr[ks], false, b1, (short)0, s1, false, false);
    }

    // Online softmax per output row (C layout: VGPR r -> row mrowoff+r, key = lane group).
#pragma unroll
    for (int r = 0; r < 8; ++r) {
      int qi  = q0 + mrowoff + r;
      int kn0 = k0 + lm;
      int kn1 = k0 + 16 + lm;
      float v0 = (kn0 <= qi) ? s0[r] * SCALE_F : -3.0e38f;
      float v1 = (kn1 <= qi) ? s1[r] * SCALE_F : -3.0e38f;
      float t = fmaxf(v0, v1);
#pragma unroll
      for (int off = 1; off < 16; off <<= 1)
        t = fmaxf(t, __shfl_xor(t, off, 16));
      float mnew = fmaxf(mrun[r], t);
      float corr = __expf(mrun[r] - mnew);
      float p0 = __expf(v0 - mnew);
      float p1 = __expf(v1 - mnew);
      float rs = p0 + p1;
#pragma unroll
      for (int off = 1; off < 16; off <<= 1)
        rs += __shfl_xor(rs, off, 16);
      lrun[r] = lrun[r] * corr + rs;
      mrun[r] = mnew;
#pragma unroll
      for (int dt = 0; dt < 8; ++dt) oacc[dt][r] *= corr;
      int mr = mrowoff + r;
      pl[mr * 32 + lm]      = f2bf(p0);
      pl[mr * 32 + 16 + lm] = f2bf(p1);
    }

    // Reload P as an A fragment (16x32) and do O += P @ V.
    const __bf16* pp = pl + lm * 32 + lk * 8;
    bf16x16 pfrag = mk_frag(pp, pp + 16);
#pragma unroll
    for (int dt = 0; dt < 8; ++dt) {
      const __bf16* vp = Vb + (size_t)(dt * 16 + lm) * SS + k0 + lk * 8;
      bf16x16 vfrag = mk_frag(vp, vp + 16);
      oacc[dt] = __builtin_amdgcn_wmma_f32_16x16x32_bf16(false, pfrag, false, vfrag, (short)0, oacc[dt], false, false);
    }
  }

  // Normalize and store: O[B*S, NH*HD]
#pragma unroll
  for (int r = 0; r < 8; ++r) {
    float inv = 1.0f / lrun[r];
    size_t row = (size_t)(b * SS + q0 + mrowoff + r) * (NH * HD) + (size_t)h * HD;
#pragma unroll
    for (int dt = 0; dt < 8; ++dt)
      O[row + dt * 16 + lm] = f2bf(oacc[dt][r] * inv);
  }
}

// ---------------- host-side orchestration ----------------

extern "C" void kernel_launch(void* const* d_in, const int* in_sizes, int n_in,
                              void* d_out, int out_size, void* d_ws, size_t ws_size,
                              hipStream_t stream) {
  const float* hs   = (const float*)d_in[0];
  const float* cosb = (const float*)d_in[1];
  const float* sinb = (const float*)d_in[2];
  // d_in[3] attention_mask: causal mask reconstructed in-kernel.
  const float* wq = (const float*)d_in[4];
  const float* wk = (const float*)d_in[5];
  const float* wv = (const float*)d_in[6];
  const float* wo = (const float*)d_in[7];
  float* out = (float*)d_out;

  char* ws = (char*)d_ws;
  size_t off = 0;
  auto alloc = [&](size_t bytes) -> void* {
    void* p = ws + off;
    off = (off + bytes + 255) & ~(size_t)255;
    return p;
  };

  __bf16* hs_bf   = (__bf16*)alloc((size_t)BSROWS * HIDD * 2);
  __bf16* wq_t    = (__bf16*)alloc((size_t)HIDD * (NH  * HD) * 2);
  __bf16* wk_t    = (__bf16*)alloc((size_t)HIDD * (NKH * HD) * 2);
  __bf16* wv_t    = (__bf16*)alloc((size_t)HIDD * (NKH * HD) * 2);
  __bf16* wo_t    = (__bf16*)alloc((size_t)(NH * HD) * HIDD * 2);
  float*  qf      = (float*) alloc((size_t)BSROWS * (NH  * HD) * 4);
  float*  kf      = (float*) alloc((size_t)BSROWS * (NKH * HD) * 4);
  float*  vf      = (float*) alloc((size_t)BSROWS * (NKH * HD) * 4);
  __bf16* q_bf    = (__bf16*)alloc((size_t)BSROWS * (NH  * HD) * 2);
  __bf16* k_bf    = (__bf16*)alloc((size_t)BSROWS * (NKH * HD) * 2);
  __bf16* v_t     = (__bf16*)alloc((size_t)BSROWS * (NKH * HD) * 2);
  __bf16* attn_bf = (__bf16*)alloc((size_t)BSROWS * (NH  * HD) * 2);

  const int T = 256;
  auto blocks = [](size_t n, int t) { return (unsigned)((n + t - 1) / t); };

  // 1) precision conversion (+ weight transposes)
  cvt_bf16_kernel<<<blocks((size_t)BSROWS * HIDD, T), T, 0, stream>>>(hs, hs_bf, BSROWS * HIDD);
  cvt_bf16_t_kernel<<<blocks((size_t)HIDD * NH  * HD, T), T, 0, stream>>>(wq, wq_t, HIDD, NH  * HD);
  cvt_bf16_t_kernel<<<blocks((size_t)HIDD * NKH * HD, T), T, 0, stream>>>(wk, wk_t, HIDD, NKH * HD);
  cvt_bf16_t_kernel<<<blocks((size_t)HIDD * NKH * HD, T), T, 0, stream>>>(wv, wv_t, HIDD, NKH * HD);
  cvt_bf16_t_kernel<<<blocks((size_t)(NH * HD) * HIDD, T), T, 0, stream>>>(wo, wo_t, NH * HD, HIDD);

  // 2) QKV projections (async-LDS double-buffered WMMA GEMMs)
  {
    dim3 gq((NH  * HD) / 128, BSROWS / 128);
    dim3 gk((NKH * HD) / 128, BSROWS / 128);
    gemm_bf16_kernel<<<gq, 256, 0, stream>>>(hs_bf, wq_t, qf, BSROWS, NH  * HD, HIDD);
    gemm_bf16_kernel<<<gk, 256, 0, stream>>>(hs_bf, wk_t, kf, BSROWS, NKH * HD, HIDD);
    gemm_bf16_kernel<<<gk, 256, 0, stream>>>(hs_bf, wv_t, vf, BSROWS, NKH * HD, HIDD);
  }

  // 3) RoPE + layout change for attention
  rope_kernel<<<blocks((size_t)BSROWS * NH  * HD, T), T, 0, stream>>>(qf, cosb, sinb, q_bf, NH);
  rope_kernel<<<blocks((size_t)BSROWS * NKH * HD, T), T, 0, stream>>>(kf, cosb, sinb, k_bf, NKH);
  v_transpose_kernel<<<blocks((size_t)BSROWS * NKH * HD, T), T, 0, stream>>>(vf, v_t);

  // 4) causal GQA flash attention (WMMA)
  {
    dim3 ga(SS / 128, NH, BB);
    flash_attn_kernel<<<ga, 256, 0, stream>>>(q_bf, k_bf, v_t, attn_bf);
  }

  // 5) output projection (WMMA GEMM) -> f32 out
  {
    dim3 go(HIDD / 128, BSROWS / 128);
    gemm_bf16_kernel<<<go, 256, 0, stream>>>(attn_bf, wo_t, out, BSROWS, HIDD, HIDD);
  }

  (void)in_sizes; (void)n_in; (void)out_size; (void)ws_size;
}
